// CNN_43284680409182
// MI455X (gfx1250) — matrix operand order repaired and checked by reading, not compiled, not run on hardware
//
#include <hip/hip_runtime.h>
#include <hip/hip_bf16.h>

typedef _Float16 f16;
typedef __attribute__((ext_vector_type(16))) _Float16 v16h;
typedef __attribute__((ext_vector_type(8)))  _Float16 v8h;
typedef __attribute__((ext_vector_type(8)))  float    v8f;
typedef __attribute__((ext_vector_type(2)))  float    v2f;
typedef __attribute__((ext_vector_type(4)))  int      v4i;

#define DEVFN __device__ __forceinline__

#ifndef __has_builtin
#define __has_builtin(x) 0
#endif

#if __has_builtin(__builtin_amdgcn_global_load_async_to_lds_b128)
#define HAVE_ASYNC_LDS 1
#else
#define HAVE_ASYNC_LDS 0
#endif

typedef __attribute__((address_space(1))) v4i* gv4i_p;
typedef __attribute__((address_space(3))) v4i* lv4i_p;

// 16-byte global -> LDS copy; async (ASYNCcnt-tracked) when available.
DEVFN void async_copy16(const f16* __restrict__ g, f16* __restrict__ l)
{
#if HAVE_ASYNC_LDS
    __builtin_amdgcn_global_load_async_to_lds_b128(
        (gv4i_p)(f16*)g, (lv4i_p)l, 0, 0);
#else
    *(v8h*)l = *(const v8h*)g;   // sync fallback: global_load_b128 + ds_store_b128
#endif
}

DEVFN void wait_async()
{
#if HAVE_ASYNC_LDS
#if __has_builtin(__builtin_amdgcn_s_wait_asynccnt)
    __builtin_amdgcn_s_wait_asynccnt((short)0);
#else
    asm volatile("s_wait_asynccnt 0x0" ::: "memory");
#endif
#endif
}

// ---------------------------------------------------------------------------
// Weight repack: OIHW fp32 -> per-lane WMMA A-fragment f16 layout.
// K ordering: gK = tap*Cin + c  (matches B fragment = NHWC channel run per tap)
// A 16x32 f16 per-lane k order chosen here (B uses the same): lane<16 (M=lane): K {0..15};
//                              lane>=16 (M=lane-16): K {16..31}
// ---------------------------------------------------------------------------
__global__ __launch_bounds__(64) void repack_weights(
    const float* __restrict__ w, f16* __restrict__ out,
    int Cout, int Cin, int taps, int mtiles, int kblocks)
{
    int tid = blockIdx.x * 64 + threadIdx.x;
    int total = mtiles * kblocks * 32;
    if (tid >= total) return;
    int lane = tid & 31;
    int blk  = tid >> 5;              // mt*kblocks + kb
    int kb   = blk % kblocks;
    int mt   = blk / kblocks;
    int m    = mt * 16 + (lane & 15);
    int half = lane >> 4;

    v16h r;
    #pragma unroll
    for (int i = 0; i < 16; ++i) {
        int K  = i + half * 16;
        int gK = kb * 32 + K;
        int tap = gK / Cin;
        int c   = gK % Cin;
        float v = 0.0f;
        if (m < Cout && tap < taps)
            v = w[((long)m * Cin + c) * taps + tap];
        r[i] = (f16)v;
    }
    *(v16h*)(out + (size_t)tid * 16) = r;
}

// ---------------------------------------------------------------------------
// conv1: 1 -> 64, 5x5, pad 2, PReLU. Direct FMA, weights+bias in LDS.
// Output: NHWC f16, channel stride 64.
// ---------------------------------------------------------------------------
__global__ __launch_bounds__(256) void conv1_direct(
    const float* __restrict__ x, const float* __restrict__ w,
    const float* __restrict__ bias, const float* __restrict__ prelu_a,
    f16* __restrict__ act_out, int Bn, int H, int W)
{
    __shared__ float wl[64 * 25];
    __shared__ float bl[64];
    for (int i = threadIdx.x; i < 64 * 25; i += 256) wl[i] = w[i];
    if (threadIdx.x < 64) bl[threadIdx.x] = bias[threadIdx.x];
    __syncthreads();

    long gid = (long)blockIdx.x * 256 + threadIdx.x;
    long P   = (long)Bn * H * W;
    if (gid >= P) return;

    int  xx = (int)(gid % W);
    long t  = gid / W;
    int  y  = (int)(t % H);
    int  b  = (int)(t / H);
    float pa = prelu_a[0];

    const float* plane = x + (long)b * H * W;
    float in[25];
    #pragma unroll
    for (int dy = 0; dy < 5; ++dy) {
        int iy = y + dy - 2;
        #pragma unroll
        for (int dx = 0; dx < 5; ++dx) {
            int ix = xx + dx - 2;
            bool v = (iy >= 0) & (iy < H) & (ix >= 0) & (ix < W);
            in[dy * 5 + dx] = v ? plane[(long)iy * W + ix] : 0.0f;
        }
    }

    f16* outp = act_out + gid * 64;
    for (int c8 = 0; c8 < 8; ++c8) {
        v8h o;
        #pragma unroll
        for (int r = 0; r < 8; ++r) {
            int c = c8 * 8 + r;
            float a = bl[c];
            #pragma unroll
            for (int tp = 0; tp < 25; ++tp)
                a = fmaf(in[tp], wl[c * 25 + tp], a);
            a = (a >= 0.0f) ? a : pa * a;
            o[r] = (f16)a;
        }
        *(v8h*)(outp + c8 * 8) = o;
    }
}

// ---------------------------------------------------------------------------
// WMMA implicit-GEMM 3x3 conv, pad 1, with async-staged LDS input slab.
// Block = 192 threads = 6 waves; each wave owns a 16-pixel x-strip, block
// covers 96 pixels of one row. The (y-1..y+1) x 98-pixel x CIN slab is
// DMA'd into LDS once (ASYNCcnt), halo zeros pre-filled only for border
// blocks, so the WMMA loop has zero masking and reads B from LDS.
// ---------------------------------------------------------------------------
template <int CIN, int MTILES, int COUT_STRIDE, bool PRELU>
__global__ __launch_bounds__(192) void conv3x3_wmma_lds(
    const f16* __restrict__ act_in, const f16* __restrict__ wpack,
    const float* __restrict__ bias, const float* __restrict__ prelu_a,
    f16* __restrict__ act_out, int Bn, int H, int W, int coutReal)
{
    constexpr int KBLK = 9 * (CIN / 32);
    constexpr int TP   = 96;          // pixels per block
    constexpr int HP   = TP + 2;      // + halo
    __shared__ alignas(32) f16 lds[3 * HP * CIN];

    const int tid  = threadIdx.x;
    const int lane = tid & 31;
    const int wv   = tid >> 5;        // 0..5
    const int npix = lane & 15;
    const int half = lane >> 4;

    const int tilesPerRow = W / TP;   // 10 (exact)
    const int x0  = ((int)blockIdx.x % tilesPerRow) * TP;
    const int row = (int)blockIdx.x / tilesPerRow;
    const int y   = row % H;
    const int b   = row / H;
    const long planeBase = (long)b * H * W;

    // ---- stage input slab into LDS ----------------------------------------
    const bool interior = (y > 0) & (y < H - 1) & (x0 > 0) & (x0 + TP < W);
    if (!interior) {                               // block-uniform branch
        for (int i = tid; i < 3 * HP * CIN / 8; i += 192)
            ((v8h*)lds)[i] = (v8h){};
        __syncthreads();
    }
    constexpr int CHUNKS = 3 * HP * (CIN / 8);     // 16-byte chunks
    for (int c = tid; c < CHUNKS; c += 192) {
        int r   = c / (HP * (CIN / 8));
        int rem = c - r * (HP * (CIN / 8));
        int pix = rem / (CIN / 8);
        int ch8 = rem - pix * (CIN / 8);
        int iy  = y + r - 1;
        int ix  = x0 - 1 + pix;
        if ((iy >= 0) & (iy < H) & (ix >= 0) & (ix < W)) {
            const f16* g = act_in + (planeBase + (long)iy * W + ix) * CIN + ch8 * 8;
            f16*       l = &lds[(r * HP + pix) * CIN + ch8 * 8];
            async_copy16(g, l);
        }
    }
    wait_async();
    __syncthreads();

    // ---- WMMA MAC loop (all data in LDS, no masking) ----------------------
    v8f acc[MTILES];
    #pragma unroll
    for (int mt = 0; mt < MTILES; ++mt) acc[mt] = (v8f){};

    const int pixL = wv * 16 + npix + 1;           // LDS pixel index (halo=+1)

    #pragma unroll
    for (int dy = 0; dy < 3; ++dy) {
        #pragma unroll
        for (int dx = 0; dx < 3; ++dx) {
            const f16* src = &lds[(dy * HP + (pixL + dx - 1)) * CIN + half * 16];
            #pragma unroll
            for (int cb = 0; cb < CIN / 32; ++cb) {
                v16h bf = *(const v16h*)(src + cb * 32);
                const int kblk = (dy * 3 + dx) * (CIN / 32) + cb;
                #pragma unroll
                for (int mt = 0; mt < MTILES; ++mt) {
                    v16h af = *(const v16h*)(wpack +
                               (size_t)((mt * KBLK + kblk) * 32 + lane) * 16);
                    acc[mt] = __builtin_amdgcn_wmma_f32_16x16x32_f16(
                        false, af, false, bf, (short)0, acc[mt], false, false);
                }
            }
        }
    }

    // ---- epilogue: bias (+PReLU), f16 pack, one b128 store per m-tile -----
    const float pa   = PRELU ? prelu_a[0] : 0.0f;
    const int   x    = x0 + wv * 16 + npix;
    const long  opix = planeBase + (long)y * W + x;
    #pragma unroll
    for (int mt = 0; mt < MTILES; ++mt) {
        const int mbase = mt * 16 + half * 8;      // 8 consecutive channels
        if (mbase + 8 > COUT_STRIDE) continue;     // conv4 padded tail
        v8h o;
        #pragma unroll
        for (int r = 0; r < 8; ++r) {
            int ch = mbase + r;
            int bi = ch < coutReal ? ch : coutReal - 1;
            float v = acc[mt][r] + bias[bi];
            if (PRELU) v = (v >= 0.0f) ? v : pa * v;
            o[r] = (f16)v;
        }
        *(v8h*)(act_out + opix * COUT_STRIDE + mbase) = o;
    }
}

// ---------------------------------------------------------------------------
// Head: per low-res pixel, 37 channels -> 4 sub-pixel outputs (2x2 shuffle).
// ---------------------------------------------------------------------------
__global__ __launch_bounds__(256) void head_kernel(
    const f16* __restrict__ act4, float* __restrict__ out,
    int Bn, int H, int W)
{
    long gid = (long)blockIdx.x * 256 + threadIdx.x;
    long P   = (long)Bn * H * W;
    if (gid >= P) return;

    int  xx = (int)(gid % W);
    long t  = gid / W;
    int  y  = (int)(t % H);
    int  b  = (int)(t / H);

    const f16* ch = act4 + gid * 40;
    float c[40];
    #pragma unroll
    for (int i = 0; i < 5; ++i) {
        v8h v = ((const v8h*)ch)[i];
        #pragma unroll
        for (int r = 0; r < 8; ++r) c[i * 8 + r] = (float)v[r];
    }

    const float cx[4] = {-0.25f, 0.25f, -0.25f, 0.25f};
    const float cy[4] = { 0.25f, 0.25f, -0.25f, -0.25f};
    float o[4];
    #pragma unroll
    for (int k = 0; k < 4; ++k) o[k] = c[36];      // shared bias channel
    #pragma unroll
    for (int j = 0; j < 12; ++j) {
        float a  = c[2 * j];
        float bb = c[2 * j + 1];
        float w1 = c[24 + j];
        #pragma unroll
        for (int k = 0; k < 4; ++k) {
            float h = cx[k] * a + cy[k] * bb;
            h = h > 0.0f ? h : 0.0f;
            o[k] = fmaf(h, w1, o[k]);
        }
    }

    // out[b, 2y+k1, 2x+k2] = o[k1*2+k2]
    const int W2 = 2 * W;
    float* obase = out + ((long)b * (2 * H) + 2 * y) * W2 + 2 * xx;
    *(v2f*)(obase)      = (v2f){o[0], o[1]};
    *(v2f*)(obase + W2) = (v2f){o[2], o[3]};
}

// ---------------------------------------------------------------------------
extern "C" void kernel_launch(void* const* d_in, const int* in_sizes, int n_in,
                              void* d_out, int out_size, void* d_ws, size_t ws_size,
                              hipStream_t stream)
{
    const float* x       = (const float*)d_in[0];
    const float* conv1_w = (const float*)d_in[1];
    const float* conv1_b = (const float*)d_in[2];
    const float* conv2_w = (const float*)d_in[3];
    const float* conv2_b = (const float*)d_in[4];
    const float* conv3_w = (const float*)d_in[5];
    const float* conv3_b = (const float*)d_in[6];
    const float* conv4_w = (const float*)d_in[7];
    const float* conv4_b = (const float*)d_in[8];
    const float* prelu_a = (const float*)d_in[9];
    float* out = (float*)d_out;

    const int Bn = 4, H = 540, W = 960;
    const long P = (long)Bn * H * W;               // 2,073,600

    // Workspace layout (bytes)
    char* ws = (char*)d_ws;
    const size_t o_w1 = 0;                               // 4*1*32*16*2   = 4096
    const size_t o_w2 = o_w1 + 4096;                     // 2*18*32*16*2  = 36864
    const size_t o_w3 = o_w2 + 36864;                    // 2*9*32*16*2   = 18432
    const size_t o_w4 = o_w3 + 18432;                    // 3*9*32*16*2   = 27648
    const size_t o_a1 = o_w4 + 27648;                    // act1/act3 region
    const size_t sz_a1 = (size_t)P * 64 * sizeof(f16);   // 265.4 MB
    const size_t o_a2 = o_a1 + sz_a1;                    // act2/act4 region (166 MB)

    f16* w1p  = (f16*)(ws + o_w1);
    f16* w2p  = (f16*)(ws + o_w2);
    f16* w3p  = (f16*)(ws + o_w3);
    f16* w4p  = (f16*)(ws + o_w4);
    f16* act1 = (f16*)(ws + o_a1);   // NHWC, C=64
    f16* act2 = (f16*)(ws + o_a2);   // NHWC, C=32
    f16* act3 = (f16*)(ws + o_a1);   // reuse act1 (dead after conv2)
    f16* act4 = (f16*)(ws + o_a2);   // reuse act2 (dead after conv4 reads act3), C-stride 40

    // Repack weights into WMMA A-fragment order (tiny launches)
    repack_weights<<<(4 * 1 * 32 + 63) / 64, 64, 0, stream>>>(conv1_w, w1p, 64, 1, 25, 4, 1);
    repack_weights<<<(2 * 18 * 32 + 63) / 64, 64, 0, stream>>>(conv2_w, w2p, 32, 64, 9, 2, 18);
    repack_weights<<<(2 * 9 * 32 + 63) / 64, 64, 0, stream>>>(conv3_w, w3p, 32, 32, 9, 2, 9);
    repack_weights<<<(3 * 9 * 32 + 63) / 64, 64, 0, stream>>>(conv4_w, w4p, 37, 32, 9, 3, 9);

    // conv1: direct (Cin=1), LDS-staged weights
    conv1_direct<<<(int)(P / 256), 256, 0, stream>>>(
        x, conv1_w, conv1_b, prelu_a, act1, Bn, H, W);

    // conv2/3/4: WMMA implicit GEMM, 96 pixels/block, 6 waves of 192 threads
    const int blocks = Bn * H * (W / 96);          // exact: 21600

    conv3x3_wmma_lds<64, 2, 32, true><<<blocks, 192, 0, stream>>>(
        act1, w2p, conv2_b, prelu_a, act2, Bn, H, W, 32);
    conv3x3_wmma_lds<32, 2, 32, true><<<blocks, 192, 0, stream>>>(
        act2, w3p, conv3_b, prelu_a, act3, Bn, H, W, 32);
    conv3x3_wmma_lds<32, 3, 40, false><<<blocks, 192, 0, stream>>>(
        act3, w4p, conv4_b, prelu_a, act4, Bn, H, W, 37);

    // Head: bilinear-weight evaluation + 2x2 pixel shuffle
    head_kernel<<<(int)(P / 256), 256, 0, stream>>>(act4, out, Bn, H, W);

    (void)in_sizes; (void)n_in; (void)out_size; (void)ws_size;
}
